// JambaRLPolicy_81827716923672
// MI455X (gfx1250) — matrix operand order repaired and checked by reading, not compiled, not run on hardware
//
#include <hip/hip_runtime.h>

typedef __attribute__((ext_vector_type(16))) __bf16 v16bf;
typedef __attribute__((ext_vector_type(8)))  __bf16 bf16x8;
typedef __attribute__((ext_vector_type(8)))  float  v8f;
typedef __attribute__((ext_vector_type(4)))  unsigned int uint32x4;
typedef __attribute__((ext_vector_type(8)))  int int32x8;
typedef __attribute__((ext_vector_type(4)))  int int32x4;

#define BM 128
#define BN 128
#define BK 32

enum { EPI_NONE = 0, EPI_TANH = 1, EPI_GELU = 2, EPI_RES = 3, EPI_SOFTPLUS = 4 };

union ABfrag { v16bf v; bf16x8 h[2]; };

__device__ __forceinline__ float gelu_f(float x) {
    float x3 = x * x * x;
    return 0.5f * x * (1.0f + tanhf(0.7978845608028654f * (x + 0.044715f * x3)));
}
__device__ __forceinline__ float softplus_f(float x) {
    return (x > 20.0f) ? x : log1pf(expf(x));
}
__device__ __forceinline__ float silu_f(float x) {
    return x / (1.0f + expf(-x));
}

// ---------------------------------------------------------------------------
// TDM: DMA one B-tile (tile_dim0=32 k-elems contiguous, tile_dim1=128 n-rows)
// of the pre-transposed bf16 weight Wt[N][K] into LDS at lds_off.
// Descriptor per CDNA5 ISA 8.3/8.4: 2D tensor, data_size=2B, OOB rows -> 0.
// ---------------------------------------------------------------------------
__device__ __forceinline__ void tdm_load_b_tile(const __bf16* Wt, int K, int N,
                                                int n0, int k0, unsigned lds_off)
{
    unsigned long long ga = (unsigned long long)(size_t)(Wt + (size_t)n0 * K + k0);
    unsigned td0 = (unsigned)(K - k0);      // remaining extent dim0 (>=32)
    unsigned td1 = (unsigned)(N - n0);      // remaining extent dim1 (clips N=96 tile)
    uint32x4 g0;
    g0[0] = 1u;                                                  // count=1, user D#
    g0[1] = lds_off;                                             // LDS byte address
    g0[2] = (unsigned)ga;                                        // global_addr[31:0]
    g0[3] = ((unsigned)(ga >> 32) & 0x01FFFFFFu) | (2u << 30);   // addr[56:32] | type=2
    int32x8 g1;
    g1[0] = (int)(1u << 16);                                     // data_size=1 -> 2 bytes
    g1[1] = (int)((td0 & 0xFFFFu) << 16);                        // tensor_dim0[15:0] @ bit48
    g1[2] = (int)((td0 >> 16) | ((td1 & 0xFFFFu) << 16));        // td0[31:16] | td1[15:0]
    g1[3] = (int)((td1 >> 16) | (32u << 16));                    // td1[31:16] | tile_dim0=32
    g1[4] = (int)(128u);                                         // tile_dim1=128, tile_dim2=0
    g1[5] = (int)(unsigned)K;                                    // tensor_dim0_stride[31:0]
    g1[6] = 0;                                                   // stride[47:32]=0, dim1_stride=0
    g1[7] = 0;
    int32x4 z4 = {0, 0, 0, 0};
    int32x8 z8 = {0, 0, 0, 0, 0, 0, 0, 0};
    __builtin_amdgcn_tensor_load_to_lds(g0, g1, z4, z4, z8, 0);
}

// ---------------------------------------------------------------------------
// Generic GEMM: C[M,N] = epi(A[M,K] @ W[K,N] (+bias) (+R))
// A fp32 (converted to bf16 while staging); W pre-transposed bf16 Wt[N][K],
// B-tiles DMA'd into LDS by the Tensor Data Mover (wave 0 issues, TENSORcnt
// pipelined against the double buffer). 8 waves; wave tile 64x32 = 4x2
// v_wmma_f32_16x16x32_bf16 accumulators. NG guards only the epilogue (TDM
// OOB clipping zero-fills the staged tile for N=96).
// ---------------------------------------------------------------------------
template <int EPI, bool NG>
__global__ __launch_bounds__(256)
void gemm_bf16_kernel(const float* __restrict__ A, int lda,
                      const __bf16* __restrict__ Wt,
                      const float* __restrict__ bias,
                      const float* __restrict__ R,
                      float* __restrict__ C, int ldc,
                      int N, int K)
{
    __shared__ alignas(16) __bf16 As[2][BM * BK];    // row-major [m][k]
    __shared__ alignas(16) __bf16 Bs[2][BN * BK];    // packed    [n][k] (TDM dest)

    const int tid   = threadIdx.x;
    const int lane  = tid & 31;
    const int wave  = tid >> 5;
    const int wm    = wave & 1;    // 0..1 -> 64 rows each
    const int wn    = wave >> 1;   // 0..3 -> 32 cols each
    const int lhalf = lane >> 4;   // 0: lanes 0-15, 1: lanes 16-31
    const int l15   = lane & 15;
    const bool issuer = (wave == 0);

    const int m0 = blockIdx.y * BM;
    const int n0 = blockIdx.x * BN;

    const v8f vzero = {0.f, 0.f, 0.f, 0.f, 0.f, 0.f, 0.f, 0.f};
    v8f acc[4][2];
#pragma unroll
    for (int m = 0; m < 4; m++)
#pragma unroll
        for (int n = 0; n < 2; n++) acc[m][n] = vzero;

    float4 ar[4];

    auto load_stage_a = [&](int k0) {
#pragma unroll
        for (int i = 0; i < 4; i++) {
            int flat = i * 1024 + tid * 4;
            ar[i] = *(const float4*)(A + (size_t)(m0 + (flat >> 5)) * lda + k0 + (flat & 31));
        }
    };
    auto store_stage_a = [&](int buf) {
#pragma unroll
        for (int i = 0; i < 4; i++) {
            int flat = i * 1024 + tid * 4;
            __bf16* d = &As[buf][(flat >> 5) * BK + (flat & 31)];
            d[0] = (__bf16)ar[i].x; d[1] = (__bf16)ar[i].y;
            d[2] = (__bf16)ar[i].z; d[3] = (__bf16)ar[i].w;
        }
    };
    auto compute = [&](int buf) {
        ABfrag a[4], b[2];
#pragma unroll
        for (int m = 0; m < 4; m++) {
            // lanes 0-15: K{0..7},{16..23}; lanes 16-31: K{8..15},{24..31}
            const __bf16* p = &As[buf][(wm * 64 + m * 16 + l15) * BK + (lhalf ? 8 : 0)];
            a[m].h[0] = *(const bf16x8*)p;
            a[m].h[1] = *(const bf16x8*)(p + 16);
        }
#pragma unroll
        for (int n = 0; n < 2; n++) {
            // lanes 0-15: K{0..7},{16..23}; lanes 16-31: K{8..15},{24..31}; column = lane&15
            const __bf16* p = &Bs[buf][(wn * 32 + n * 16 + l15) * BK + lhalf * 8];
            b[n].h[0] = *(const bf16x8*)p;
            b[n].h[1] = *(const bf16x8*)(p + 16);
        }
#pragma unroll
        for (int m = 0; m < 4; m++)
#pragma unroll
            for (int n = 0; n < 2; n++)
                acc[m][n] = __builtin_amdgcn_wmma_f32_16x16x32_bf16(
                    false, a[m].v, false, b[n].v, (short)0, acc[m][n], false, false);
    };

    // ---- software pipeline: TDM(B,next) + global(A,next) overlap compute ----
    if (issuer) tdm_load_b_tile(Wt, K, N, n0, 0, (unsigned)(size_t)&Bs[0][0]);
    load_stage_a(0);
    store_stage_a(0);
    if (issuer) __builtin_amdgcn_s_wait_tensorcnt(0);
    __syncthreads();
    int cur = 0;
    for (int k0 = 0; k0 < K; k0 += BK) {
        const bool has_next = (k0 + BK) < K;
        if (has_next) {
            load_stage_a(k0 + BK);                 // global loads in flight
            if (issuer)
                tdm_load_b_tile(Wt, K, N, n0, k0 + BK,
                                (unsigned)(size_t)&Bs[cur ^ 1][0]);
        }
        compute(cur);
        if (has_next) {
            store_stage_a(cur ^ 1);
            if (issuer) __builtin_amdgcn_s_wait_tensorcnt(0);
            cur ^= 1;
        }
        __syncthreads();
    }

    // ---- epilogue + store (C/D layout: VGPR r -> M = r + lhalf*8, N = lane&15) ----
#pragma unroll
    for (int m = 0; m < 4; m++) {
#pragma unroll
        for (int n = 0; n < 2; n++) {
            int col = n0 + wn * 32 + n * 16 + l15;
            if (NG && col >= N) continue;
            int rbase = m0 + wm * 64 + m * 16 + lhalf * 8;
            float bv = bias ? bias[col] : 0.0f;
#pragma unroll
            for (int r = 0; r < 8; r++) {
                int row = rbase + r;
                float v = acc[m][n][r] + bv;
                if (EPI == EPI_TANH)          v = tanhf(v);
                else if (EPI == EPI_GELU)     v = gelu_f(v);
                else if (EPI == EPI_SOFTPLUS) v = softplus_f(v);
                else if (EPI == EPI_RES)      v += R[(size_t)row * ldc + col];
                C[(size_t)row * ldc + col] = v;
            }
        }
    }
}

// ---------------------------------------------------------------------------
// Weight prep: Wt[N][K] bf16 = transpose(W[K][N] fp32). 32x32 LDS tiles,
// coalesced reads and writes. Grid (N/32, K/32), 256 threads.
// ---------------------------------------------------------------------------
__global__ __launch_bounds__(256)
void transpose_cvt_kernel(const float* __restrict__ W, __bf16* __restrict__ Wt,
                          int K, int N)
{
    __shared__ float tile[32][33];
    const int n0 = blockIdx.x * 32, k0 = blockIdx.y * 32;
    const int tc = threadIdx.x & 31;
    const int tr = threadIdx.x >> 5;      // 8 rows per pass
#pragma unroll
    for (int i = 0; i < 4; i++) {
        int k = tr + i * 8;
        tile[k][tc] = W[(size_t)(k0 + k) * N + n0 + tc];
    }
    __syncthreads();
#pragma unroll
    for (int i = 0; i < 4; i++) {
        int n = tr + i * 8;
        Wt[(size_t)(n0 + n) * K + k0 + tc] = (__bf16)tile[tc][n];
    }
}

// ---------------------------------------------------------------------------
// Row LayerNorm over 1024 columns; one 256-thread block per row.
// ---------------------------------------------------------------------------
__global__ __launch_bounds__(256)
void ln_kernel(const float* __restrict__ in, const float* __restrict__ g,
               const float* __restrict__ b, float* __restrict__ out,
               int do_tanh)
{
    __shared__ float red[256];
    const int cols = 1024;
    int row = blockIdx.x;
    const float* x = in + (size_t)row * cols;
    float v[4];
    float s = 0.f;
#pragma unroll
    for (int i = 0; i < 4; i++) { v[i] = x[threadIdx.x + i * 256]; s += v[i]; }
    red[threadIdx.x] = s; __syncthreads();
    for (int off = 128; off > 0; off >>= 1) {
        if (threadIdx.x < off) red[threadIdx.x] += red[threadIdx.x + off];
        __syncthreads();
    }
    float mean = red[0] / cols;
    __syncthreads();
    float vs = 0.f;
#pragma unroll
    for (int i = 0; i < 4; i++) { float d = v[i] - mean; vs += d * d; }
    red[threadIdx.x] = vs; __syncthreads();
    for (int off = 128; off > 0; off >>= 1) {
        if (threadIdx.x < off) red[threadIdx.x] += red[threadIdx.x + off];
        __syncthreads();
    }
    float inv = rsqrtf(red[0] / cols + 1e-5f);
#pragma unroll
    for (int i = 0; i < 4; i++) {
        int c = threadIdx.x + i * 256;
        float o = (v[i] - mean) * inv * g[c] + b[c];
        if (do_tanh) o = tanhf(o);
        out[(size_t)row * cols + c] = o;
    }
}

// ---------------------------------------------------------------------------
// Mamba front: x = silu(x * conv_w[:, -1] + conv_b)  (x = first half of xz)
// ---------------------------------------------------------------------------
__global__ __launch_bounds__(256)
void mamba_front_kernel(const float* __restrict__ xz,
                        const float* __restrict__ conv_w,
                        const float* __restrict__ conv_b,
                        float* __restrict__ xs)
{
    int idx = blockIdx.x * 256 + threadIdx.x;   // = b*2048 + d
    int b = idx >> 11, d = idx & 2047;
    float x = xz[(size_t)b * 4096 + d];
    x = x * conv_w[d * 4 + 3] + conv_b[d];
    xs[idx] = silu_f(x);
}

// ---------------------------------------------------------------------------
// Fused SSM scan step (bandwidth-critical; explicit b128 traffic):
//   dA = exp(dt * -exp(A_log)); ns = state*dA + (dt*x)*Bm; y = sum(ns*Cm)
//   y = (y + D*x) * silu(z).  Writes ns straight into d_out.
// ---------------------------------------------------------------------------
__global__ __launch_bounds__(256)
void mamba_scan_kernel(const float* __restrict__ xs,
                       const float* __restrict__ xz,
                       const float* __restrict__ dt,
                       const float* __restrict__ dbl,
                       const float* __restrict__ A_log,
                       const float* __restrict__ Dp,
                       const float* __restrict__ state,
                       float* __restrict__ ns,
                       float* __restrict__ y)
{
    int idx = blockIdx.x * 256 + threadIdx.x;   // = b*2048 + d
    int b = idx >> 11, d = idx & 2047;
    float dtv = dt[idx];
    float xv  = xs[idx];
    float zv  = xz[(size_t)b * 4096 + 2048 + d];
    const float4* bm4 = (const float4*)(dbl + (size_t)b * 96 + 64);   // dbl[:, 64:80]
    const float4* cm4 = (const float4*)(dbl + (size_t)b * 96 + 80);   // dbl[:, 80:96]
    const float4* al4 = (const float4*)(A_log + (size_t)d * 16);
    const float4* st4 = (const float4*)(state + (size_t)idx * 16);
    float4*       no4 = (float4*)(ns + (size_t)idx * 16);

    float4 stv[4], alv[4], bmv[4], cmv[4];
#pragma unroll
    for (int i = 0; i < 4; i++) {
        stv[i] = st4[i]; alv[i] = al4[i]; bmv[i] = bm4[i]; cmv[i] = cm4[i];
    }
    float accv = 0.f;
    float dx = dtv * xv;
#pragma unroll
    for (int i = 0; i < 4; i++) {
        float4 r;
        r.x = stv[i].x * expf(dtv * -expf(alv[i].x)) + dx * bmv[i].x;
        r.y = stv[i].y * expf(dtv * -expf(alv[i].y)) + dx * bmv[i].y;
        r.z = stv[i].z * expf(dtv * -expf(alv[i].z)) + dx * bmv[i].z;
        r.w = stv[i].w * expf(dtv * -expf(alv[i].w)) + dx * bmv[i].w;
        accv += r.x * cmv[i].x + r.y * cmv[i].y + r.z * cmv[i].z + r.w * cmv[i].w;
        no4[i] = r;
    }
    y[idx] = (accv + Dp[d] * xv) * silu_f(zv);
}

// ---------------------------------------------------------------------------
// Host-side dispatch
// ---------------------------------------------------------------------------
static void run_gemm(int epi, const float* A, int lda, const __bf16* Wt,
                     const float* bias, const float* R, float* C, int ldc,
                     int M, int N, int K, hipStream_t st)
{
    dim3 grid((N + BN - 1) / BN, (M + BM - 1) / BM), blk(256);
    const bool ng = (N % BN) != 0;
#define GEMM_CASE(E)                                                                   \
    case E:                                                                            \
        if (ng) gemm_bf16_kernel<E, true ><<<grid, blk, 0, st>>>(A, lda, Wt, bias, R, C, ldc, N, K); \
        else    gemm_bf16_kernel<E, false><<<grid, blk, 0, st>>>(A, lda, Wt, bias, R, C, ldc, N, K); \
        break;
    switch (epi) {
        GEMM_CASE(EPI_NONE)
        GEMM_CASE(EPI_TANH)
        GEMM_CASE(EPI_GELU)
        GEMM_CASE(EPI_RES)
        GEMM_CASE(EPI_SOFTPLUS)
    }
#undef GEMM_CASE
}

extern "C" void kernel_launch(void* const* d_in, const int* in_sizes, int n_in,
                              void* d_out, int out_size, void* d_ws, size_t ws_size,
                              hipStream_t stream)
{
    (void)in_sizes; (void)n_in; (void)out_size; (void)ws_size;
    const int Bsz = 2048, H = 1024, DI = 2048, MLPH = 4096, ADIM = 128, OBS = 512;

    auto in = [&](int i) { return (const float*)d_in[i]; };
    // jax tree (sorted-dict) flatten order:
    const float* obs      = in(0);
    const float* fin_b    = in(1);
    const float* fin_g    = in(2);
    const float* inp_b    = in(3);
    const float* inp_ln_b = in(4);
    const float* inp_ln_g = in(5);
    const float* inp_w    = in(6);
    const int L0 = 7, LSZ = 17, ATT = 58;
    const float* out1_b = in(71);
    const float* out1_w = in(72);
    const float* out2_b = in(73);
    const float* out2_w = in(74);
    const float* states[3] = { in(75), in(76), in(77) };

    float* outF = (float*)d_out;
    float* actions = outF;
    const size_t ACT_N = (size_t)Bsz * ADIM;            // 262144
    const size_t ST_N  = (size_t)Bsz * DI * 16;         // 67108864
    float* nsOut[3] = { outF + ACT_N, outF + ACT_N + ST_N, outF + ACT_N + 2 * ST_N };

    float* ws = (float*)d_ws;
    size_t off = 0;
    auto alloc = [&](size_t n) { float* p = ws + off; off += n; return p; };
    float* h   = alloc((size_t)Bsz * H);
    float* x   = alloc((size_t)Bsz * H);
    float* xz  = alloc((size_t)Bsz * 2 * DI);   // reused as MLP hidden
    float* xs  = alloc((size_t)Bsz * DI);
    float* dbl = alloc((size_t)Bsz * 96);
    float* dt  = alloc((size_t)Bsz * DI);
    float* y   = alloc((size_t)Bsz * DI);
    float* h2  = alloc((size_t)Bsz * H);
    float* t   = xz;

    // bf16 transposed-weight arena after the fp32 scratch
    __bf16* warena = (__bf16*)(ws + off);
    __bf16* wptr = warena;
    auto prep_w = [&](const float* W, int K, int N) -> const __bf16* {
        __bf16* dst = wptr; wptr += (size_t)K * N;
        dim3 g(N / 32, K / 32), b(256);
        transpose_cvt_kernel<<<g, b, 0, stream>>>(W, dst, K, N);
        return dst;
    };

    // ---- one pass of weight transposition/conversion (per launch) ----
    const __bf16* inp_wt = prep_w(inp_w, OBS, H);
    const __bf16 *in_wt[3], *xp_wt[3], *dt_wt[3], *out_wt[3];
    const __bf16 *mlp_w1t[4], *mlp_w2t[4];
    for (int li = 0; li < 3; li++) {
        int base = L0 + li * LSZ;
        in_wt[li]   = prep_w(in(base + 6), H, 2 * DI);
        xp_wt[li]   = prep_w(in(base + 16), DI, 96);
        dt_wt[li]   = prep_w(in(base + 5), 64, DI);
        out_wt[li]  = prep_w(in(base + 15), DI, H);
        mlp_w1t[li] = prep_w(in(base + 9), H, MLPH);
        mlp_w2t[li] = prep_w(in(base + 10), MLPH, H);
    }
    const __bf16* wv_t = prep_w(in(ATT + 12), H, H);
    const __bf16* wo_t = prep_w(in(ATT + 10), H, H);
    mlp_w1t[3] = prep_w(in(ATT + 3), H, MLPH);
    mlp_w2t[3] = prep_w(in(ATT + 4), MLPH, H);
    const __bf16* out1_wt = prep_w(out1_w, H, H);
    const __bf16* out2_wt = prep_w(out2_w, H, ADIM);

    dim3 blk(256);
    const int EW = (Bsz * DI) / 256;   // elementwise grid over (b,d)

    // input projection + LN + tanh
    run_gemm(EPI_NONE, obs, OBS, inp_wt, inp_b, nullptr, x, H, Bsz, H, OBS, stream);
    ln_kernel<<<Bsz, blk, 0, stream>>>(x, inp_ln_g, inp_ln_b, h, 1);

    int si = 0;
    for (int li = 0; li < 4; li++) {
        bool isAttn = (li == 3);
        int base = isAttn ? ATT : (L0 + li * LSZ);
        if (!isAttn) {
            const float* A_log  = in(base + 0);
            const float* Dp     = in(base + 1);
            const float* conv_b = in(base + 2);
            const float* conv_w = in(base + 3);
            const float* dt_b   = in(base + 4);
            const float* mlp_b1 = in(base + 7);
            const float* mlp_b2 = in(base + 8);
            const float* n1_b   = in(base + 11);
            const float* n1_g   = in(base + 12);
            const float* n2_b   = in(base + 13);
            const float* n2_g   = in(base + 14);

            ln_kernel<<<Bsz, blk, 0, stream>>>(h, n1_g, n1_b, x, 0);
            run_gemm(EPI_NONE, x, H, in_wt[li], nullptr, nullptr, xz, 2 * DI, Bsz, 2 * DI, H, stream);
            mamba_front_kernel<<<EW, blk, 0, stream>>>(xz, conv_w, conv_b, xs);
            run_gemm(EPI_NONE, xs, DI, xp_wt[li], nullptr, nullptr, dbl, 96, Bsz, 96, DI, stream);
            run_gemm(EPI_SOFTPLUS, dbl, 96, dt_wt[li], dt_b, nullptr, dt, DI, Bsz, DI, 64, stream);
            mamba_scan_kernel<<<EW, blk, 0, stream>>>(xs, xz, dt, dbl, A_log, Dp,
                                                      states[si], nsOut[si], y);
            run_gemm(EPI_RES, y, DI, out_wt[li], nullptr, h, h, H, Bsz, H, DI, stream);
            si++;

            ln_kernel<<<Bsz, blk, 0, stream>>>(h, n2_g, n2_b, x, 0);
            run_gemm(EPI_GELU, x, H, mlp_w1t[li], mlp_b1, nullptr, t, MLPH, Bsz, MLPH, H, stream);
            run_gemm(EPI_RES, t, MLPH, mlp_w2t[li], mlp_b2, h, h, H, Bsz, H, MLPH, stream);
        } else {
            const float* bo     = in(base + 0);
            const float* mlp_b1 = in(base + 1);
            const float* mlp_b2 = in(base + 2);
            const float* n1_b   = in(base + 5);
            const float* n1_g   = in(base + 6);
            const float* n2_b   = in(base + 7);
            const float* n2_g   = in(base + 8);

            ln_kernel<<<Bsz, blk, 0, stream>>>(h, n1_g, n1_b, x, 0);
            run_gemm(EPI_NONE, x, H, wv_t, nullptr, nullptr, y, H, Bsz, H, H, stream);
            run_gemm(EPI_RES, y, H, wo_t, bo, h, h, H, Bsz, H, H, stream);

            ln_kernel<<<Bsz, blk, 0, stream>>>(h, n2_g, n2_b, x, 0);
            run_gemm(EPI_GELU, x, H, mlp_w1t[3], mlp_b1, nullptr, t, MLPH, Bsz, MLPH, H, stream);
            run_gemm(EPI_RES, t, MLPH, mlp_w2t[3], mlp_b2, h, h, H, Bsz, H, MLPH, stream);
        }
    }

    // final head
    ln_kernel<<<Bsz, blk, 0, stream>>>(h, fin_g, fin_b, x, 0);
    run_gemm(EPI_TANH, x, H, out1_wt, out1_b, nullptr, h2, H, Bsz, H, H, stream);
    run_gemm(EPI_TANH, h2, H, out2_wt, out2_b, nullptr, actions, ADIM, Bsz, ADIM, H, stream);
}
